// ProtoNetCLS_28561532518728
// MI455X (gfx1250) — matrix operand order repaired and checked by reading, not compiled, not run on hardware
//
#include <hip/hip_runtime.h>
#include <hip/hip_bf16.h>
#include <math.h>
#include <stdint.h>

// Problem dims (fixed by reference)
#define B_IMG   64
#define HWXY    256
#define NPATCH  16384      // B*HW
#define DB      768
#define DIM     256
#define NC      201
#define NK      5
#define CKTOT   1005       // C*K
#define CKPAD   1024

typedef __attribute__((ext_vector_type(16))) __bf16 v16bf;
typedef __attribute__((ext_vector_type(8)))  float  v8f;

#if defined(__HIP_DEVICE_COMPILE__) && \
    __has_builtin(__builtin_amdgcn_tensor_load_to_lds) && \
    __has_builtin(__builtin_amdgcn_s_wait_tensorcnt)
#define USE_TDM 1
#else
#define USE_TDM 0
#endif

#if USE_TDM
typedef __attribute__((ext_vector_type(4))) unsigned int u32x4;
typedef __attribute__((ext_vector_type(8))) int          i32x8;
typedef __attribute__((ext_vector_type(4))) int          i32x4;

// Generic pointers into LDS carry the LDS byte offset in their low 32 bits
// (aperture bits live in [63:32] per CDNA5 flat addressing), which is what
// the D#.lds_addr field wants.
__device__ __forceinline__ unsigned lds_off(const void* p) {
  return (unsigned)(uintptr_t)p;
}

// Issue one 2-D TDM tile load: tile_d1 rows x tile_d0 f32 elements, row
// stride stride_d0 (elements), packed contiguously into LDS at ldsaddr.
// D# layout per CDNA5 ISA ch.8: group0 = {count/type, lds_addr, global_addr},
// group1 = {data_size, tensor dims/strides, tile dims}. 2-D tiles leave the
// upper descriptor groups zero (clang-23 6-arg form: g2, g3, extra, cpol).
__device__ __forceinline__ void tdm_load_2d(const void* gaddr, unsigned ldsaddr,
                                            unsigned tensor_d0, unsigned tensor_d1,
                                            unsigned stride_d0,
                                            unsigned tile_d0, unsigned tile_d1) {
  const unsigned long long ga = (unsigned long long)(uintptr_t)gaddr;
  u32x4 g0;
  g0[0] = 1u;                                               // count=1 (valid), user mode
  g0[1] = ldsaddr;                                          // lds_addr      [63:32]
  g0[2] = (unsigned)(ga & 0xFFFFFFFFu);                     // global_addr   [95:64]
  g0[3] = (unsigned)((ga >> 32) & 0x01FFFFFFu) | (2u << 30);// addr[56:32] | type=2
  i32x8 g1;
  g1[0] = (int)(2u << 16);                                  // data_size = 4B
  g1[1] = (int)((tensor_d0 & 0xFFFFu) << 16);               // tensor_dim0[15:0]
  g1[2] = (int)(((tensor_d0 >> 16) & 0xFFFFu) |
                ((tensor_d1 & 0xFFFFu) << 16));             // dim0 hi | dim1 lo
  g1[3] = (int)(((tensor_d1 >> 16) & 0xFFFFu) |
                ((tile_d0 & 0xFFFFu) << 16));               // dim1 hi | tile_dim0
  g1[4] = (int)(tile_d1 & 0xFFFFu);                         // tile_dim1, tile_dim2=0
  g1[5] = (int)stride_d0;                                   // tensor_dim0_stride lo
  g1[6] = 0;                                                // stride hi | dim1_stride lo
  g1[7] = 0;
  const i32x4 gz4 = {0, 0, 0, 0};
  const i32x8 gz8 = {0, 0, 0, 0, 0, 0, 0, 0};
  __builtin_amdgcn_tensor_load_to_lds(g0, g1, gz4, gz4, gz8, 0);
}
#endif

// ---------------------------------------------------------------- helpers ---

__device__ __forceinline__ __bf16 f2bf(float f) {
  unsigned u = __builtin_bit_cast(unsigned, f);
  u += 0x7FFFu + ((u >> 16) & 1u);           // round to nearest even
  unsigned short h = (unsigned short)(u >> 16);
  return __builtin_bit_cast(__bf16, h);
}

// A fragment: 16(M) x 32(K) bf16 from row-major f32 (leading dim ld).
// Contiguous k order used here (B matches): lanes 0-15 carry K 0-15, lanes 16-31 K 16-31.
__device__ __forceinline__ v16bf load_a_frag(const float* src, int ld, int lane) {
  const int m  = lane & 15;
  const int kb = (lane >> 4) << 4;           // 0 or 16
  const float* row = src + (size_t)m * ld;
  v16bf a;
#pragma unroll
  for (int e = 0; e < 16; ++e) {
    const int k = kb + e;
    a[e] = f2bf(row[k]);
  }
  return a;
}

// B fragment: 32(K) x 16(N) bf16 where B[k][n] = W[n*ld + k] (W row-major N x K).
// Lanes 0-15: K 0-15, lanes 16-31: K 16-31.
__device__ __forceinline__ v16bf load_b_fragT(const float* W, int ld, int lane) {
  const int n  = lane & 15;
  const int kb = (lane >> 4) << 4;           // 0 or 16
  const float* col = W + (size_t)n * ld + kb;
  v16bf b;
#pragma unroll
  for (int e = 0; e < 16; ++e) b[e] = f2bf(col[e]);
  return b;
}

// ---------------------------------------------------- K1: l2n(prototypes) ---
__global__ void __launch_bounds__(256) k_proto_norm(const float* __restrict__ prot,
                                                    float* __restrict__ protosP) {
  __shared__ float red[256];
  const int ck = blockIdx.x, t = threadIdx.x;
  float v = (ck < CKTOT) ? prot[(size_t)ck * DIM + t] : 0.f;
  red[t] = v * v; __syncthreads();
  for (int s = 128; s > 0; s >>= 1) { if (t < s) red[t] += red[t + s]; __syncthreads(); }
  const float inv = 1.f / fmaxf(sqrtf(red[0]), 1e-12f);
  protosP[(size_t)ck * DIM + t] = v * inv;
}

// ------------------------------- K2: proj GEMM + bias + LayerNorm + l2norm ---
// Block = 8 waves, tile = 16 rows x 256 cols (full D) so LN/l2n fuse in LDS.
// A/B tiles staged into LDS by the Tensor Data Mover each K-step.
__global__ void __launch_bounds__(256) k_proj_ln(const float* __restrict__ tok,
                                                 const float* __restrict__ w,
                                                 const float* __restrict__ bias,
                                                 const float* __restrict__ g,
                                                 const float* __restrict__ bb,
                                                 float* __restrict__ patches) {
  __shared__ float smA[16 * 32];     // A tile (rows x K)
  __shared__ float smB[DIM * 32];    // B tile (N x K)
  __shared__ float smO[16 * DIM];    // output tile for fused LN/l2n
  __shared__ float redA[256];
  __shared__ float redB[256];
  const int t = threadIdx.x;
  const int wv = t >> 5, lane = t & 31;
  const int rbase = blockIdx.x * 16;
  const int n0 = wv * 32, n1 = n0 + 16;
  v8f acc0 = {}, acc1 = {};
#pragma unroll 1
  for (int kt = 0; kt < DB; kt += 32) {
#if USE_TDM
    if (wv == 0) {
      tdm_load_2d(tok + (size_t)rbase * DB + kt, lds_off(smA), DB, NPATCH, DB, 32, 16);
      tdm_load_2d(w + kt,                        lds_off(smB), DB, DIM,    DB, 32, DIM);
      __builtin_amdgcn_s_wait_tensorcnt(0);
    }
#else
    if (kt + 32 < DB) __builtin_prefetch(tok + (size_t)rbase * DB + kt + 32, 0, 1);
    for (int i = t; i < 16 * 32; i += 256)
      smA[i] = tok[(size_t)(rbase + (i >> 5)) * DB + kt + (i & 31)];
    for (int i = t; i < DIM * 32; i += 256)
      smB[i] = w[(size_t)(i >> 5) * DB + kt + (i & 31)];
#endif
    __syncthreads();
    v16bf a  = load_a_frag(smA, 32, lane);
    v16bf b0 = load_b_fragT(smB + n0 * 32, 32, lane);
    v16bf b1 = load_b_fragT(smB + n1 * 32, 32, lane);
    acc0 = __builtin_amdgcn_wmma_f32_16x16x32_bf16(false, a, false, b0, (short)0, acc0, false, false);
    acc1 = __builtin_amdgcn_wmma_f32_16x16x32_bf16(false, a, false, b1, (short)0, acc1, false, false);
    __syncthreads();
  }
  { // C layout: VGPR r -> row r (lanes 0-15) / row 8+r (lanes 16-31), col = lane&15
    const int n = lane & 15;
    const int mo = (lane >> 4) ? 8 : 0;
#pragma unroll
    for (int r = 0; r < 8; ++r) {
      smO[(mo + r) * DIM + n0 + n] = acc0[r] + bias[n0 + n];
      smO[(mo + r) * DIM + n1 + n] = acc1[r] + bias[n1 + n];
    }
  }
  __syncthreads();
  // LayerNorm + l2norm: 16 threads per row, 16 cols per thread
  const int row = t >> 4, seg = t & 15;
  float x[16];
  float s1 = 0.f, s2 = 0.f;
#pragma unroll
  for (int i = 0; i < 16; ++i) {
    x[i] = smO[row * DIM + seg * 16 + i];
    s1 += x[i]; s2 += x[i] * x[i];
  }
  redA[t] = s1; redB[t] = s2; __syncthreads();
  for (int s = 8; s > 0; s >>= 1) {
    if (seg < s) { redA[t] += redA[t + s]; redB[t] += redB[t + s]; }
    __syncthreads();
  }
  const float mu  = redA[row * 16] * (1.f / DIM);
  const float var = redB[row * 16] * (1.f / DIM) - mu * mu;
  const float rsd = rsqrtf(var + 1e-5f);
  float yss = 0.f;
#pragma unroll
  for (int i = 0; i < 16; ++i) {
    const int col = seg * 16 + i;
    x[i] = (x[i] - mu) * rsd * g[col] + bb[col];
    yss += x[i] * x[i];
  }
  __syncthreads();
  redA[t] = yss; __syncthreads();
  for (int s = 8; s > 0; s >>= 1) { if (seg < s) redA[t] += redA[t + s]; __syncthreads(); }
  const float inv = 1.f / fmaxf(sqrtf(redA[row * 16]), 1e-12f);
#pragma unroll
  for (int i = 0; i < 16; ++i)
    patches[(size_t)(rbase + row) * DIM + seg * 16 + i] = x[i] * inv;
}

// --------------------------- K3: logits GEMM, written straight into d_out ---
// Block = 8 waves as 2(M) x 4(N); tile 32 x 128; N padded to 1024, store-masked.
__global__ void __launch_bounds__(256) k_logits(const float* __restrict__ patches,
                                                const float* __restrict__ protosP,
                                                float* __restrict__ logits) {
  __shared__ float smA[32 * 32];     // A tile (32 rows x K32)
  __shared__ float smB[128 * 32];    // B tile (128 n x K32)
  const int t = threadIdx.x, wv = t >> 5, lane = t & 31;
  const int wm = wv >> 2, wn = wv & 3;
  const int rblk = blockIdx.x * 32;
  const int nblk = blockIdx.y * 128;
  v8f acc0 = {}, acc1 = {};
#pragma unroll 1
  for (int kt = 0; kt < DIM; kt += 32) {
#if USE_TDM
    if (wv == 0) {
      tdm_load_2d(patches + (size_t)rblk * DIM + kt, lds_off(smA), DIM, NPATCH, DIM, 32, 32);
      tdm_load_2d(protosP + (size_t)nblk * DIM + kt, lds_off(smB), DIM, CKPAD,  DIM, 32, 128);
      __builtin_amdgcn_s_wait_tensorcnt(0);
    }
#else
    for (int i = t; i < 32 * 32; i += 256)
      smA[i] = patches[(size_t)(rblk + (i >> 5)) * DIM + kt + (i & 31)];
    for (int i = t; i < 128 * 32; i += 256)
      smB[i] = protosP[(size_t)(nblk + (i >> 5)) * DIM + kt + (i & 31)];
#endif
    __syncthreads();
    v16bf a  = load_a_frag(smA + (wm * 16) * 32, 32, lane);
    v16bf b0 = load_b_fragT(smB + (wn * 32) * 32, 32, lane);
    v16bf b1 = load_b_fragT(smB + (wn * 32 + 16) * 32, 32, lane);
    acc0 = __builtin_amdgcn_wmma_f32_16x16x32_bf16(false, a, false, b0, (short)0, acc0, false, false);
    acc1 = __builtin_amdgcn_wmma_f32_16x16x32_bf16(false, a, false, b1, (short)0, acc1, false, false);
    __syncthreads();
  }
  const int n = lane & 15, mo = (lane >> 4) ? 8 : 0;
  const int rbase = rblk + wm * 16;
  const int nb = nblk + wn * 32;
#pragma unroll
  for (int r = 0; r < 8; ++r) {
    const int row = rbase + mo + r;
    const int c0 = nb + n, c1 = nb + 16 + n;
    if (c0 < CKTOT) logits[(size_t)row * CKTOT + c0] = acc0[r];
    if (c1 < CKTOT) logits[(size_t)row * CKTOT + c1] = acc1[r];
  }
}

// -------------------- K4: class_preds (argmax_c max_k) -> correct_f flags ---
__global__ void __launch_bounds__(64) k_classpred(const float* __restrict__ logits,
                                                  const int* __restrict__ lbl,
                                                  float* __restrict__ correct) {
  __shared__ float rv[64]; __shared__ int ri[64];
  const int np = blockIdx.x, t = threadIdx.x;
  const float* lrow = logits + (size_t)np * CKTOT;
  float bv = -3.402823466e38f; int bi = NC;
  for (int c = t; c < NC; c += 64) {
    float m = lrow[c * NK];
#pragma unroll
    for (int k = 1; k < NK; ++k) m = fmaxf(m, lrow[c * NK + k]);
    if (m > bv || (m == bv && c < bi)) { bv = m; bi = c; }
  }
  rv[t] = bv; ri[t] = bi; __syncthreads();
  for (int s = 32; s > 0; s >>= 1) {
    if (t < s) {
      if (rv[t + s] > rv[t] || (rv[t + s] == rv[t] && ri[t + s] < ri[t])) {
        rv[t] = rv[t + s]; ri[t] = ri[t + s];
      }
    }
    __syncthreads();
  }
  if (t == 0) correct[np] = (lbl[np] == ri[0]) ? 1.f : 0.f;
}

// --------------------------- K5: ppl transpose + img_logits (mean over HW) ---
__global__ void __launch_bounds__(256) k_ppl(const float* __restrict__ logits,
                                             float* __restrict__ ppl,
                                             float* __restrict__ img) {
  __shared__ float red[256];
  const int ck = blockIdx.x, b = blockIdx.y, t = threadIdx.x;
  const float v = logits[((size_t)b * HWXY + t) * CKTOT + ck];
  ppl[((size_t)b * CKTOT + ck) * HWXY + t] = v;
  red[t] = v; __syncthreads();
  for (int s = 128; s > 0; s >>= 1) { if (t < s) red[t] += red[t + s]; __syncthreads(); }
  if (t == 0) img[(size_t)b * CKTOT + ck] = red[0] * (1.f / HWXY);
}

// --------------------------------------------- K6: pred_logits (sum over k) ---
__global__ void k_pred(const float* __restrict__ img, float* __restrict__ pred) {
  const int i = blockIdx.x * blockDim.x + threadIdx.x;
  if (i >= B_IMG * NC) return;
  const int b = i / NC, c = i % NC;
  float s = 0.f;
#pragma unroll
  for (int k = 0; k < NK; ++k) s += img[(size_t)b * CKTOT + c * NK + k];
  pred[i] = s;
}

// ---------------- K7: per-class masked Sinkhorn (implicit L) + top-scores ---
__global__ void __launch_bounds__(256) k_sinkhorn(const float* __restrict__ logits,
                                                  const int* __restrict__ lbl,
                                                  const float* __restrict__ correct,
                                                  float* __restrict__ target,
                                                  int* __restrict__ topIdx,
                                                  int* __restrict__ has) {
  __shared__ float f[NK];
  __shared__ float red[256 * NK];
  __shared__ float rv[256]; __shared__ int ri[256];
  __shared__ int cnt;
  const int c = blockIdx.x, t = threadIdx.x;
  if (t == 0) cnt = 0;
  if (t < NK) f[t] = 1.f;
  __syncthreads();
  int myCnt = 0;
  for (int it = 0; it < 3; ++it) {
    float fl[NK];
#pragma unroll
    for (int k = 0; k < NK; ++k) fl[k] = f[k];
    float cs[NK] = {0.f, 0.f, 0.f, 0.f, 0.f};
    for (int n = t; n < NPATCH; n += 256) {
      if (lbl[n] != c) continue;
      if (it == 0) myCnt++;
      const float* lr = logits + (size_t)n * CKTOT + c * NK;
      float E[NK];
#pragma unroll
      for (int k = 0; k < NK; ++k) E[k] = __expf(lr[k] * 20.0f);  // 1/SK_EPS
      float r = 1.f;
      if (it > 0) {
        float rsum = 0.f;
#pragma unroll
        for (int k = 0; k < NK; ++k) rsum += E[k] * fl[k];
        r = 1.f / fmaxf(rsum, 1e-12f);
      }
#pragma unroll
      for (int k = 0; k < NK; ++k) cs[k] += E[k] * r;
    }
#pragma unroll
    for (int k = 0; k < NK; ++k) red[t * NK + k] = cs[k];
    __syncthreads();
    for (int s = 128; s > 0; s >>= 1) {
      if (t < s) {
#pragma unroll
        for (int k = 0; k < NK; ++k) red[t * NK + k] += red[(t + s) * NK + k];
      }
      __syncthreads();
    }
    if (t == 0) {
#pragma unroll
      for (int k = 0; k < NK; ++k) {
        const float csk = f[k] * red[k];                  // actual column sum of L
        f[k] = f[k] / (fmaxf(csk, 1e-12f) * (float)NK);
      }
    }
    __syncthreads();
  }
  atomicAdd(&cnt, myCnt);
  // Final pass: hard assignment + per-k argmax over score (zeros included).
  float fl[NK];
#pragma unroll
  for (int k = 0; k < NK; ++k) fl[k] = f[k];
  float bv[NK]; int bn[NK];
#pragma unroll
  for (int k = 0; k < NK; ++k) { bv[k] = -3.402823466e38f; bn[k] = NPATCH; }
  for (int n = t; n < NPATCH; n += 256) {
    float s[NK] = {0.f, 0.f, 0.f, 0.f, 0.f};
    if (lbl[n] == c) {
      const float* lr = logits + (size_t)n * CKTOT + c * NK;
      float l[NK];
      float best = -3.402823466e38f; int bidx = 0;
#pragma unroll
      for (int k = 0; k < NK; ++k) {
        l[k] = lr[k];
        const float Ek = __expf(l[k] * 20.0f) * fl[k];
        if (Ek > best) { best = Ek; bidx = k; }           // first-k on ties
      }
      target[n] = (float)(bidx + NK * c);
      if (correct[n] > 0.f) s[bidx] = l[bidx];
    }
#pragma unroll
    for (int k = 0; k < NK; ++k)
      if (s[k] > bv[k] || (s[k] == bv[k] && n < bn[k])) { bv[k] = s[k]; bn[k] = n; }
  }
  __syncthreads();
  for (int k = 0; k < NK; ++k) {
    rv[t] = bv[k]; ri[t] = bn[k]; __syncthreads();
    for (int s = 128; s > 0; s >>= 1) {
      if (t < s) {
        if (rv[t + s] > rv[t] || (rv[t + s] == rv[t] && ri[t + s] < ri[t])) {
          rv[t] = rv[t + s]; ri[t] = ri[t + s];
        }
      }
      __syncthreads();
    }
    if (t == 0) topIdx[c * NK + k] = ri[0];
    __syncthreads();
  }
  if (t == 0) has[c] = (cnt > 0) ? 1 : 0;
}

// --------------------------------- K8: prototype EMA update + final l2norm ---
__global__ void __launch_bounds__(256) k_update(const float* __restrict__ patches,
                                                const float* __restrict__ correct,
                                                const float* __restrict__ protosP,
                                                const int* __restrict__ topIdx,
                                                const int* __restrict__ has,
                                                float* __restrict__ outP) {
  __shared__ float red[256];
  const int ck = blockIdx.x, t = threadIdx.x;
  const int ti = topIdx[ck];
  const float cf = correct[ti];
  float pn = patches[(size_t)ti * DIM + t] * cf;
  red[t] = pn * pn; __syncthreads();
  for (int s = 128; s > 0; s >>= 1) { if (t < s) red[t] += red[t + s]; __syncthreads(); }
  pn *= 1.f / fmaxf(sqrtf(red[0]), 1e-12f);
  const float base = protosP[(size_t)ck * DIM + t];
  const float u = has[ck / NK] ? (0.999f * base + 0.001f * pn) : base;
  __syncthreads();
  red[t] = u * u; __syncthreads();
  for (int s = 128; s > 0; s >>= 1) { if (t < s) red[t] += red[t + s]; __syncthreads(); }
  outP[(size_t)ck * DIM + t] = u / fmaxf(sqrtf(red[0]), 1e-12f);
}

// ------------------------------------------------------------------- launch ---
extern "C" void kernel_launch(void* const* d_in, const int* in_sizes, int n_in,
                              void* d_out, int out_size, void* d_ws, size_t ws_size,
                              hipStream_t stream) {
  const float* tok  = (const float*)d_in[0];   // (64,256,768)
  const float* pw   = (const float*)d_in[1];   // (256,768)
  const float* pb   = (const float*)d_in[2];   // (256,)
  const float* lg   = (const float*)d_in[3];   // (256,)
  const float* lb   = (const float*)d_in[4];   // (256,)
  const float* prot = (const float*)d_in[5];   // (201,5,256)
  const int*   lbl  = (const int*)d_in[6];     // (16384,)

  float* out    = (float*)d_out;
  float* pred   = out;                         // 64*201
  float* ppl    = pred + (size_t)B_IMG * NC;   // 64*1005*256
  float* img    = ppl  + (size_t)B_IMG * CKTOT * HWXY;  // 64*1005
  float* clog   = img  + (size_t)B_IMG * CKTOT;         // 16384*1005 (== contrast_logits)
  float* ctgt   = clog + (size_t)NPATCH * CKTOT;        // 16384
  float* nproto = ctgt + NPATCH;                        // 201*5*256

  char* ws = (char*)d_ws;
  float* patches = (float*)ws;  ws += (size_t)NPATCH * DIM * sizeof(float);
  float* protosP = (float*)ws;  ws += (size_t)CKPAD * DIM * sizeof(float);
  float* correct = (float*)ws;  ws += (size_t)NPATCH * sizeof(float);
  int*   topIdx  = (int*)ws;    ws += (size_t)CKTOT * sizeof(int);
  int*   hasF    = (int*)ws;    ws += (size_t)NC * sizeof(int);

  k_proto_norm<<<CKPAD, 256, 0, stream>>>(prot, protosP);
  k_proj_ln  <<<NPATCH / 16, 256, 0, stream>>>(tok, pw, pb, lg, lb, patches);
  k_logits   <<<dim3(NPATCH / 32, CKPAD / 128), 256, 0, stream>>>(patches, protosP, clog);
  k_classpred<<<NPATCH, 64, 0, stream>>>(clog, lbl, correct);
  k_ppl      <<<dim3(CKTOT, B_IMG), 256, 0, stream>>>(clog, ppl, img);
  k_pred     <<<(B_IMG * NC + 255) / 256, 256, 0, stream>>>(img, pred);
  k_sinkhorn <<<NC, 256, 0, stream>>>(clog, lbl, correct, ctgt, topIdx, hasF);
  k_update   <<<CKTOT, 256, 0, stream>>>(patches, correct, protosP, topIdx, hasF, nproto);
}
